// QCO_2d_33002528702625
// MI455X (gfx1250) — matrix operand order repaired and checked by reading, not compiled, not run on hardware
//
#include <hip/hip_runtime.h>
#include <math.h>

typedef __attribute__((ext_vector_type(16))) _Float16 v16h;
typedef __attribute__((ext_vector_type(8)))  float    v8f;

#define NB 4
#define CIN 16
#define HH 256
#define WW 256
#define HWP (HH*WW)
#define LQ 16
#define INV2S2 32.0f      /* 1/(2*sigma^2), sigma = 2/L = 0.125 */
#define BN_EPS_C 1e-5f
#define EPS_C 1e-6f
#define THETA_C 0.5f

/* ---------------- Stage 1: mu = mean over H,W per (n,c) ---------------- */
__global__ void k_mu_sum(const float* __restrict__ x, float* __restrict__ musum) {
    __shared__ float sm[256];
    int nc = blockIdx.x, tid = threadIdx.x;
    const float* p = x + (size_t)nc * HWP;
    float s = 0.f;
    for (int i = tid; i < HWP; i += 256) s += p[i];
    sm[tid] = s; __syncthreads();
    for (int o = 128; o > 0; o >>= 1) { if (tid < o) sm[tid] += sm[tid + o]; __syncthreads(); }
    if (tid == 0) musum[nc] = sm[0];
}

__global__ void k_mu_norm(const float* __restrict__ musum,
                          float* __restrict__ mu, float* __restrict__ mn) {
    __shared__ float sq[64];
    int t = threadIdx.x;                       // 64 threads: (n,c)
    float m = musum[t] * (1.0f / (float)HWP);
    mu[t] = m;
    sq[t] = m * m; __syncthreads();
    int n = t >> 4;
    float s = 0.f;
    for (int c = 0; c < CIN; c++) s += sq[n * CIN + c];
    mn[t] = m / fmaxf(sqrtf(s), 1e-12f);
}

/* ---------------- Stage 2: cos map + per-block min/max ---------------- */
__global__ void k_cos(const float* __restrict__ x, const float* __restrict__ mn,
                      float* __restrict__ cosb, float* __restrict__ cpart) {
    __shared__ float smin[256], smax[256];
    int tid = threadIdx.x;
    int gid = blockIdx.x * 256 + tid;
    int n = gid >> 16, p = gid & (HWP - 1);
    const float* xp = x + (((size_t)n * CIN) << 16) + p;
    float dot = 0.f, ss = 0.f;
    for (int c = 0; c < CIN; c++) {
        float v = xp[(size_t)c << 16];
        dot += v * mn[n * CIN + c];
        ss  += v * v;
    }
    float cv = dot / fmaxf(sqrtf(ss), 1e-12f);
    cosb[gid] = cv;
    smin[tid] = cv; smax[tid] = cv; __syncthreads();
    for (int o = 128; o > 0; o >>= 1) {
        if (tid < o) { smin[tid] = fminf(smin[tid], smin[tid + o]);
                       smax[tid] = fmaxf(smax[tid], smax[tid + o]); }
        __syncthreads();
    }
    if (tid == 0) { cpart[blockIdx.x] = smin[0]; cpart[1024 + blockIdx.x] = smax[0]; }
}

/* ---------------- Stage 3: reduce min/max, build q[n][16] ---------------- */
__global__ void k_q(const float* __restrict__ cpart, float* __restrict__ q) {
    __shared__ float smin[256], smax[256];
    int tid = threadIdx.x;
    for (int n = 0; n < NB; n++) {
        smin[tid] = cpart[n * 256 + tid];
        smax[tid] = cpart[1024 + n * 256 + tid];
        __syncthreads();
        for (int o = 128; o > 0; o >>= 1) {
            if (tid < o) { smin[tid] = fminf(smin[tid], smin[tid + o]);
                           smax[tid] = fmaxf(smax[tid], smax[tid + o]); }
            __syncthreads();
        }
        if (tid == 0) {
            float cmin = smin[0], cmax = smax[0];
            for (int l = 0; l < LQ; l++)
                q[n * LQ + l] = (2.f * l + 1.f) / (2.f * LQ) * (cmax - cmin) + cmin;
        }
        __syncthreads();
    }
}

/* ------- Stage 4: global co min/max (factorized per-pixel products) ------- */
__global__ void k_mm(const float* __restrict__ cosb, const float* __restrict__ q,
                     float* __restrict__ mmpart) {
    __shared__ float smin[256], smax[256];
    int tid = threadIdx.x;
    int gid = blockIdx.x * 256 + tid;
    int n = gid >> 16, p = gid & (HWP - 1);
    int h = p >> 8, w = p & 255;
    float cv = cosb[gid];
    float dmin = 1e30f, dmax = 0.f;
    for (int l = 0; l < LQ; l++) {
        float d = cv - q[n * LQ + l]; d *= d;
        dmin = fminf(dmin, d); dmax = fmaxf(dmax, d);
    }
    float maxL = __expf(-INV2S2 * dmin), minL = __expf(-INV2S2 * dmax);
    float minR = 0.f, maxR = 0.f;
    if (h < HH - 1 && w < WW - 1) {
        float cr = cosb[gid + WW + 1];
        float rmin = 1e30f, rmax = 0.f;
        for (int l = 0; l < LQ; l++) {
            float d = cr - q[n * LQ + l]; d *= d;
            rmin = fminf(rmin, d); rmax = fmaxf(rmax, d);
        }
        maxR = __expf(-INV2S2 * rmin); minR = __expf(-INV2S2 * rmax);
    }
    smin[tid] = minL * minR; smax[tid] = maxL * maxR; __syncthreads();
    for (int o = 128; o > 0; o >>= 1) {
        if (tid < o) { smin[tid] = fminf(smin[tid], smin[tid + o]);
                       smax[tid] = fmaxf(smax[tid], smax[tid + o]); }
        __syncthreads();
    }
    if (tid == 0) { mmpart[blockIdx.x] = smin[0]; mmpart[1024 + blockIdx.x] = smax[0]; }
}

__global__ void k_mmred(const float* __restrict__ mmpart, float* __restrict__ cc) {
    __shared__ float smin[256], smax[256];
    int tid = threadIdx.x;
    float a = 1e30f, b = -1e30f;
    for (int i = tid; i < 1024; i += 256) {
        a = fminf(a, mmpart[i]); b = fmaxf(b, mmpart[1024 + i]);
    }
    smin[tid] = a; smax[tid] = b; __syncthreads();
    for (int o = 128; o > 0; o >>= 1) {
        if (tid < o) { smin[tid] = fminf(smin[tid], smin[tid + o]);
                       smax[tid] = fmaxf(smax[tid], smax[tid + o]); }
        __syncthreads();
    }
    if (tid == 0) { cc[0] = smin[0]; cc[1] = smax[0]; }
}

/* ---- Stage 5: WMMA GEMM: sta_raw[n,l,l'] = sum_p left[l,p]*right[l',p] ----
 * 16x16 output, K = 65536 per sample, v_wmma_f32_16x16x32_f16.
 * Fragments are generated in-register: each lane loads one cos of the
 * 32-wide K tile, __shfl gathers the tile-local K values each lane needs per
 * the CDNA5 16-bit A(16x32)/B(32x16) VGPR layouts, exp applied in-lane.
 * Shifted-right load is BRANCHLESS: clamped address + v_cndmask select
 * (padded positions -> 1e30 so exp(-32*inf) == 0 exactly).
 * 64 waves per sample split K; 8-VGPR f32 partials stored per wave.        */
__device__ __forceinline__ v8f wmma_tile(const float* __restrict__ cn, int pos,
                                         float ql, int laneA, int laneB, v8f acc) {
    float cl = cn[pos];
    /* branchless shifted load: h<255 && w<255 else padded-zero sentinel */
    bool valid = (pos < (HH - 1) * WW) && ((pos & (WW - 1)) != (WW - 1));
    int  off   = valid ? (WW + 1) : 0;          /* always in-range address   */
    float crv  = cn[pos + off];
    float cr   = valid ? crv : 1e30f;           /* exp(-32*inf) == 0         */
    v16h a, b;
#pragma unroll
    for (int e = 0; e < 16; e++) {
        int kA = ((e >> 3) << 4) | (e & 7) | laneA;  /* A: 16-bit 16x32 layout */
        int kB = ((e >> 3) << 4) | (e & 7) | laneA;  /* B: 16-bit 32x16 layout */
        float cA = __shfl(cl, kA, 32);
        float dA = cA - ql;
        a[e] = (_Float16)__expf(-INV2S2 * dA * dA);
        float cB = __shfl(cr, kB, 32);
        float dB = cB - ql;
        b[e] = (_Float16)__expf(-INV2S2 * dB * dB);
    }
    return __builtin_amdgcn_wmma_f32_16x16x32_f16(
        /*neg_a=*/false, a, /*neg_b=*/false, b,
        /*c_mod=*/(short)0, acc, /*reuse_a=*/false, /*reuse_b=*/false);
}

__global__ void k_wmma(const float* __restrict__ cosb, const float* __restrict__ q,
                       float* __restrict__ part) {
    int tid  = threadIdx.x;
    int lane = tid & 31, wid = tid >> 5;
    int n  = blockIdx.x >> 3;                 // 8 blocks (64 waves) per sample
    int wn = (blockIdx.x & 7) * 8 + wid;      // wave index within sample: 0..63
    const float* cn = cosb + ((size_t)n << 16);
    float ql = q[n * LQ + (lane & 15)];       // row l (A) == col l' (B) for this lane
    int laneA = (lane & 16) >> 1;             // upper half lanes: +8 on K (A layout)
    int laneB = (lane & 16);                  // upper half lanes: +16; unused, B also uses laneA
    v8f acc = {};
    int p0 = (wn << 10) + lane;               // 1024 pixels per wave = 32 K-tiles
    /* unroll x2: interleave fragment VALU of tile t+1 with WMMA of tile t */
    for (int t = 0; t < 32; t += 2) {
        int posA = p0 + (t << 5);
        __builtin_prefetch(cn + posA + 128, 0, 0);        /* tile pair after next */
        __builtin_prefetch(cn + posA + 128 + WW + 1, 0, 0);
        acc = wmma_tile(cn, posA,      ql, laneA, laneB, acc);
        acc = wmma_tile(cn, posA + 32, ql, laneA, laneB, acc);
    }
    /* f32 C/D layout: VGPR r -> M = r + (lane>=16 ? 8 : 0), N = lane&15 */
    float* op = part + ((((size_t)n * 64) + wn) << 8);
    int Ncol = lane & 15;
    int Mbase = (lane & 16) >> 1;
#pragma unroll
    for (int r = 0; r < 8; r++) op[(r + Mbase) * 16 + Ncol] = acc[r];
}

/* ------- Stage 6: reduce wave partials + global co-normalization ------- */
__global__ void k_sta(const float* __restrict__ part, const float* __restrict__ cc,
                      float* __restrict__ sta) {
    int n = blockIdx.x, t = threadIdx.x;   // t = l*16 + l'
    float s = 0.f;
    const float* p = part + (((size_t)n * 64) << 8) + t;
    for (int k = 0; k < 64; k++) s += p[(size_t)k << 8];
    float comin = cc[0], comax = cc[1];
    sta[n * 256 + t] = (s - (float)HWP * comin) / (comax - comin + EPS_C);
}

/* ------- Stage 7: head MLP: feat -> f1(leaky) -> f2(relu) -> o1 ------- */
__global__ void k_head1(const float* __restrict__ q, const float* __restrict__ sta,
                        const float* __restrict__ mu,
                        const float* __restrict__ f1w, const float* __restrict__ f1b,
                        const float* __restrict__ f2w, const float* __restrict__ f2b,
                        const float* __restrict__ o1w, const float* __restrict__ o1b,
                        float* __restrict__ t1) {
    int P = blockIdx.x * 256 + threadIdx.x;     // 0..1023
    int n = P >> 8, p = P & 255;
    float f0  = q[n * LQ + (p & 15)];           // q_h
    float f1v = q[n * LQ + (p >> 4)];           // q_w
    float f2v = sta[P];
    float h1[64];
    for (int o = 0; o < 64; o++) {
        float v = f1w[o * 3] * f0 + f1w[o * 3 + 1] * f1v + f1w[o * 3 + 2] * f2v + f1b[o];
        h1[o] = v > 0.f ? v : 0.01f * v;
    }
    float h2[128];
    for (int o = 0; o < 128; o++) {
        float v = f2b[o];
        for (int k = 0; k < 64; k++) v += f2w[o * 64 + k] * h1[k];
        h2[o] = fmaxf(v, 0.f);
    }
    for (int o = 0; o < 128; o++) {
        float v = o1b[o];
        const float* wr = o1w + o * 144;
        for (int c = 0; c < CIN; c++) v += wr[c] * mu[n * CIN + c];      // xav concat
        for (int j = 0; j < 128; j++) v += wr[16 + j] * h2[j];
        t1[(size_t)P * 128 + o] = v;
    }
}

/* BN stats over (N, positions) = 1024 samples per channel -> scale/shift */
__global__ void k_bnstat(const float* __restrict__ t, const float* __restrict__ g,
                         const float* __restrict__ b, float* __restrict__ out) {
    int o = threadIdx.x;                        // 128 threads
    float s = 0.f, s2 = 0.f;
    for (int p = 0; p < 1024; p++) {
        float v = t[(size_t)p * 128 + o];
        s += v; s2 += v * v;
    }
    float m   = s  * (1.f / 1024.f);
    float var = s2 * (1.f / 1024.f) - m * m;
    float sc  = g[o] / sqrtf(var + BN_EPS_C);
    out[o] = sc; out[128 + o] = b[o] - m * sc;
}

/* Stage 8: relu(bn1(t1)) -> o2 conv */
__global__ void k_head2(const float* __restrict__ t1, const float* __restrict__ bn1,
                        const float* __restrict__ o2w, const float* __restrict__ o2b,
                        float* __restrict__ t2) {
    int P = blockIdx.x * 256 + threadIdx.x;
    float v1[128];
    for (int j = 0; j < 128; j++) {
        float v = bn1[j] * t1[(size_t)P * 128 + j] + bn1[128 + j];
        v1[j] = fmaxf(v, 0.f);
    }
    for (int o = 0; o < 128; o++) {
        float v = o2b[o];
        for (int j = 0; j < 128; j++) v += o2w[o * 128 + j] * v1[j];
        t2[(size_t)P * 128 + o] = v;
    }
}

/* Stage 9: s[n,o] = bn2( mean over positions of t2 ) */
__global__ void k_smean(const float* __restrict__ t2, const float* __restrict__ bn2,
                        float* __restrict__ s) {
    int t = threadIdx.x;                        // 512 threads: n*128+o
    int n = t >> 7, o = t & 127;
    float sum = 0.f;
    for (int p = 0; p < 256; p++) sum += t2[((size_t)(n * 256 + p)) * 128 + o];
    s[t] = bn2[o] * (sum * (1.f / 256.f)) + bn2[128 + o];
}

/* Stage 10: scalar epilogue + global normalization */
__global__ void k_final(const float* __restrict__ s, float* __restrict__ out) {
    __shared__ float sm[512];
    int t = threadIdx.x;
    float v = s[t];
    sm[t] = v; __syncthreads();
    for (int o = 256; o > 0; o >>= 1) { if (t < o) sm[t] = fmaxf(sm[t], sm[t + o]); __syncthreads(); }
    float mx = sm[0]; __syncthreads();
    float th = THETA_C * mx;
    sm[t] = fmaxf(v - th, 0.f); __syncthreads();
    for (int o = 256; o > 0; o >>= 1) { if (t < o) sm[t] += sm[t + o]; __syncthreads(); }
    float r = sm[0] * (1.f / (float)LQ); __syncthreads();
    float de = (v > th) ? (th + r) : (v + r);
    sm[t] = de; __syncthreads();
    for (int o = 256; o > 0; o >>= 1) { if (t < o) sm[t] = fminf(sm[t], sm[t + o]); __syncthreads(); }
    float dmin = sm[0]; __syncthreads();
    sm[t] = de; __syncthreads();
    for (int o = 256; o > 0; o >>= 1) { if (t < o) sm[t] = fmaxf(sm[t], sm[t + o]); __syncthreads(); }
    float dmax = sm[0];
    out[t] = (de - dmin) / (dmax - dmin + EPS_C);
}

extern "C" void kernel_launch(void* const* d_in, const int* in_sizes, int n_in,
                              void* d_out, int out_size, void* d_ws, size_t ws_size,
                              hipStream_t stream) {
    (void)in_sizes; (void)n_in; (void)out_size; (void)ws_size;
    const float* x    = (const float*)d_in[0];
    const float* f1w  = (const float*)d_in[1];
    const float* f1b  = (const float*)d_in[2];
    const float* f2w  = (const float*)d_in[3];
    const float* f2b  = (const float*)d_in[4];
    const float* o1w  = (const float*)d_in[5];
    const float* o1b  = (const float*)d_in[6];
    const float* bn1g = (const float*)d_in[7];
    const float* bn1b = (const float*)d_in[8];
    const float* o2w  = (const float*)d_in[9];
    const float* o2b  = (const float*)d_in[10];
    const float* bn2g = (const float*)d_in[11];
    const float* bn2b = (const float*)d_in[12];
    float* out = (float*)d_out;
    float* ws = (float*)d_ws;

    /* workspace layout (float offsets) */
    float* musum  = ws + 0;          /* 64  */
    float* mu     = ws + 64;         /* 64  */
    float* mn     = ws + 128;        /* 64  */
    float* q      = ws + 192;        /* 64  */
    float* cc     = ws + 256;        /* 2: comin, comax */
    float* cosb   = ws + 512;        /* 262144 */
    float* cpart  = ws + 262656;     /* 2048   */
    float* mmpart = ws + 264704;     /* 2048   */
    float* part   = ws + 266752;     /* 65536  */
    float* sta    = ws + 332288;     /* 1024   */
    float* sbuf   = ws + 333312;     /* 512    */
    float* bn1    = ws + 333824;     /* 256    */
    float* bn2    = ws + 334080;     /* 256    */
    float* t1     = ws + 334336;     /* 131072 */
    float* t2     = ws + 465408;     /* 131072 */

    k_mu_sum <<<64,   256, 0, stream>>>(x, musum);
    k_mu_norm<<<1,    64,  0, stream>>>(musum, mu, mn);
    k_cos    <<<1024, 256, 0, stream>>>(x, mn, cosb, cpart);
    k_q      <<<1,    256, 0, stream>>>(cpart, q);
    k_mm     <<<1024, 256, 0, stream>>>(cosb, q, mmpart);
    k_mmred  <<<1,    256, 0, stream>>>(mmpart, cc);
    k_wmma   <<<32,   256, 0, stream>>>(cosb, q, part);
    k_sta    <<<4,    256, 0, stream>>>(part, cc, sta);
    k_head1  <<<4,    256, 0, stream>>>(q, sta, mu, f1w, f1b, f2w, f2b, o1w, o1b, t1);
    k_bnstat <<<1,    128, 0, stream>>>(t1, bn1g, bn1b, bn1);
    k_head2  <<<4,    256, 0, stream>>>(t1, bn1, o2w, o2b, t2);
    k_bnstat <<<1,    128, 0, stream>>>(t2, bn2g, bn2b, bn2);
    k_smean  <<<1,    512, 0, stream>>>(t2, bn2, sbuf);
    k_final  <<<1,    512, 0, stream>>>(sbuf, out);
}
